// TemporalSplitterLayer_81535659148085
// MI455X (gfx1250) — matrix operand order repaired and checked by reading, not compiled, not run on hardware
//
#include <hip/hip_runtime.h>

typedef __attribute__((ext_vector_type(16))) _Float16 v16h;
typedef __attribute__((ext_vector_type(8)))  _Float16 v8h;
typedef __attribute__((ext_vector_type(8)))  float    v8f;

union U16h { v16h v; v8h h[2]; };

constexpr int kDim  = 256;
constexpr int kFfn  = 512;
constexpr int kSeq  = 4096;
constexpr int kTok  = 16384;           // B*N
constexpr float kEps = 1e-5f;

// ---------------------------------------------------------------- WMMA helpers
static __device__ __forceinline__ v8f wmma_f16(v16h a, v16h b, v8f c) {
  // D = A(16x32 f16) * B(32x16 f16) + C(16x16 f32)
  return __builtin_amdgcn_wmma_f32_16x16x32_f16(
      /*neg_a=*/false, a, /*neg_b=*/false, b,
      /*c_mod=*/(short)0, c, /*reuse_a=*/false, /*reuse_b=*/false);
}

// A operand, 16x32 f16 tile, row-major source (lane m holds row m).
// Lanes 0-15: K = {kofs+0..7, kofs+16..23}; lanes 16-31: K = {kofs+8..15, kofs+24..31}.
static __device__ __forceinline__ v16h load_a_p(const _Float16* base, int ld, int kofs) {
  int lane = threadIdx.x & 31;
  int hi   = lane >> 4;
  const _Float16* p = base + (size_t)(lane & 15) * ld + kofs + hi * 8;
  U16h u;
  u.h[0] = *(const v8h*)(p);
  u.h[1] = *(const v8h*)(p + 16);
  return u.v;
}

// B operand, 32x16 f16 tile; source is row-major B^T (lane n holds column n of B).
// Lanes 0-15: K = {kofs+0..7, kofs+16..23}; lanes 16-31: K = {kofs+8..15, kofs+24..31}.
static __device__ __forceinline__ v16h load_b_p(const _Float16* base, int ld, int kofs) {
  int lane = threadIdx.x & 31;
  int hi   = lane >> 4;
  const _Float16* p = base + (size_t)(lane & 15) * ld + kofs + hi * 8;
  U16h u;
  u.h[0] = *(const v8h*)(p);
  u.h[1] = *(const v8h*)(p + 16);
  return u.v;
}

// Async global -> LDS, 16 bytes per lane, tracked by ASYNCcnt.
static __device__ __forceinline__ void async_copy_b128(_Float16* lds, const _Float16* g) {
  unsigned int       l  = (unsigned int)(uintptr_t)lds;          // low 32b = LDS offset
  unsigned long long ga = (unsigned long long)(uintptr_t)g;
  asm volatile("global_load_async_to_lds_b128 %0, %1, off"
               :: "v"(l), "v"(ga) : "memory");
}

// ---------------------------------------------------- weight transpose f32->f16
// wt[n*K + k] = (f16) w[k*N + n]
__global__ __launch_bounds__(256)
void k_transpose_h(const float* __restrict__ w, _Float16* __restrict__ wt,
                   int K, int N) {
  int idx = blockIdx.x * 256 + threadIdx.x;
  if (idx >= K * N) return;
  int n = idx / K, k = idx - n * K;
  wt[(size_t)n * K + k] = (_Float16)w[(size_t)k * N + n];
}

// ------------------------------------------------------------------- LayerNorm
// One wave (32 lanes) per 256-element row; each lane owns 8 columns.
template <int OUT_HALF>
__global__ __launch_bounds__(256)
void k_layernorm(const float* __restrict__ x, const float* __restrict__ gg,
                 const float* __restrict__ bb, void* __restrict__ outp, int rows) {
  int wave = threadIdx.x >> 5;
  int lane = threadIdx.x & 31;
  int row  = blockIdx.x * 8 + wave;
  if (row >= rows) return;
  const float* xr = x + (size_t)row * kDim + lane * 8;
  float4 v0 = *(const float4*)(xr);
  float4 v1 = *(const float4*)(xr + 4);
  float s = v0.x + v0.y + v0.z + v0.w + v1.x + v1.y + v1.z + v1.w;
  float q = v0.x*v0.x + v0.y*v0.y + v0.z*v0.z + v0.w*v0.w +
            v1.x*v1.x + v1.y*v1.y + v1.z*v1.z + v1.w*v1.w;
  for (int off = 16; off >= 1; off >>= 1) {
    s += __shfl_xor(s, off, 32);
    q += __shfl_xor(q, off, 32);
  }
  float mu  = s * (1.0f / kDim);
  float var = q * (1.0f / kDim) - mu * mu;
  float rs  = rsqrtf(var + kEps);
  int c = lane * 8;
  float4 g0 = *(const float4*)(gg + c), g1 = *(const float4*)(gg + c + 4);
  float4 b0 = *(const float4*)(bb + c), b1 = *(const float4*)(bb + c + 4);
  float o[8];
  o[0] = (v0.x - mu) * rs * g0.x + b0.x;  o[1] = (v0.y - mu) * rs * g0.y + b0.y;
  o[2] = (v0.z - mu) * rs * g0.z + b0.z;  o[3] = (v0.w - mu) * rs * g0.w + b0.w;
  o[4] = (v1.x - mu) * rs * g1.x + b1.x;  o[5] = (v1.y - mu) * rs * g1.y + b1.y;
  o[6] = (v1.z - mu) * rs * g1.z + b1.z;  o[7] = (v1.w - mu) * rs * g1.w + b1.w;
  if (OUT_HALF) {
    v8h h;
    for (int i = 0; i < 8; ++i) h[i] = (_Float16)o[i];
    *(v8h*)((_Float16*)outp + (size_t)row * kDim + c) = h;
  } else {
    float* op = (float*)outp + (size_t)row * kDim + c;
    *(float4*)(op)     = make_float4(o[0], o[1], o[2], o[3]);
    *(float4*)(op + 4) = make_float4(o[4], o[5], o[6], o[7]);
  }
}

// ------------------------------------------------------------------ WMMA GEMM
// C[m0:+16, n0:+16] = A[M,K] @ Bt[N,K]^T, one wave per 16x16 tile.
// MODE 0: f16 out = (acc + bias[n]) * scale            (Q / K proj)
// MODE 1: f16 out = relu(acc + bias[n])                (FFN1)
// MODE 2: f16 out = acc + bias[m]                      (V^T proj, bias by row)
// MODE 3: f32 out = acc + bias[n] + resid[m*ld + n]    (FFN2 + residual)
template <int MODE>
__global__ __launch_bounds__(32)
void k_gemm_wmma(const _Float16* __restrict__ A, const _Float16* __restrict__ Bt,
                 const float* __restrict__ bias, const float* __restrict__ resid,
                 void* __restrict__ outp, int Kd, int ldA, int ldB, int ldOut,
                 float scale) {
  int n0 = blockIdx.x * 16;
  int m0 = blockIdx.y * 16;
  int lane = threadIdx.x & 31;
  int hi = lane >> 4, lx = lane & 15;

  v8f acc = {};
  const _Float16* Ab = A  + (size_t)m0 * ldA;
  const _Float16* Bb = Bt + (size_t)n0 * ldB;
  for (int kc = 0; kc < Kd; kc += 32) {
    v16h a = load_a_p(Ab, ldA, kc);
    v16h b = load_b_p(Bb, ldB, kc);
    acc = wmma_f16(a, b, acc);
  }
#pragma unroll
  for (int g = 0; g < 8; ++g) {
    int m = m0 + g + 8 * hi;
    int n = n0 + lx;
    float v = acc[g];
    if (MODE == 0) {
      v = (v + bias[n]) * scale;
      ((_Float16*)outp)[(size_t)m * ldOut + n] = (_Float16)v;
    } else if (MODE == 1) {
      v = fmaxf(v + bias[n], 0.0f);
      ((_Float16*)outp)[(size_t)m * ldOut + n] = (_Float16)v;
    } else if (MODE == 2) {
      v = v + bias[m];
      ((_Float16*)outp)[(size_t)m * ldOut + n] = (_Float16)v;
    } else {
      v = v + bias[n] + resid[(size_t)m * ldOut + n];
      ((float*)outp)[(size_t)m * ldOut + n] = v;
    }
  }
}

// --------------------------------------------------------- fused attention
// 4 waves / block; each wave owns a 16-query tile (64 queries per block, all in
// one batch since 4096 % 64 == 0). K and V^T tiles for each 32-key step are
// staged into LDS once per block via async global->LDS copies (double-buffered,
// ASYNCcnt-tracked) and shared by all 4 waves: 4x less L2 traffic than private
// streaming. O = relu(Qs K^T) V accumulated flash-style; R = x + O.
constexpr int ATTW  = 4;    // waves per block
constexpr int KT_LD = 264;  // 256 + 8 pad halves: 528B row, 16B aligned, conflict-free
constexpr int VT_LD = 56;   // 32 + 24 pad halves: 112B row, 16B aligned, conflict-free
constexpr int ST_LD = 56;

__global__ __launch_bounds__(128)
void k_attention(const _Float16* __restrict__ Q, const _Float16* __restrict__ Kf,
                 const _Float16* __restrict__ Vt, const float* __restrict__ x,
                 float* __restrict__ Rf, _Float16* __restrict__ Rh) {
  __shared__ __align__(16) _Float16 kt[2][32 * KT_LD];   // [key 0..31][d 0..255]
  __shared__ __align__(16) _Float16 vt[2][256 * VT_LD];  // [d 0..255][key 0..31]
  __shared__ __align__(16) _Float16 st[ATTW][16 * ST_LD];

  const int tid  = threadIdx.x;
  const int wave = tid >> 5;
  const int lane = tid & 31;
  const int hi = lane >> 4, lx = lane & 15;
  const int t0    = (blockIdx.x * ATTW + wave) * 16;   // query token base (this wave)
  const int kbase = ((blockIdx.x * ATTW * 16) >> 12) << 12;  // batch key base

  // issue one 32-key K/V tile pair into LDS buffer `buf` (16B/lane/op, ASYNCcnt)
  auto issue = [&](int jb, int buf) {
    int k0 = kbase + jb * 32;
#pragma unroll
    for (int i = 0; i < 8; ++i) {                // K tile: 32 x 256 halves
      int ci = tid * 8 + i;                      // 0..1023 chunks of 8 halves
      int r = ci >> 5, c = (ci & 31) * 8;
      async_copy_b128(&kt[buf][r * KT_LD + c], Kf + (size_t)(k0 + r) * kDim + c);
    }
#pragma unroll
    for (int i = 0; i < 8; ++i) {                // V^T tile: 256 x 32 halves
      int ci = tid * 8 + i;
      int d = ci >> 2, c = (ci & 3) * 8;
      async_copy_b128(&vt[buf][d * VT_LD + c], Vt + (size_t)d * kTok + k0 + c);
    }
  };

  // Q tile resident in registers: 8 K-chunks of 32.
  v16h aq[8];
#pragma unroll
  for (int kc = 0; kc < 8; ++kc)
    aq[kc] = load_a_p(Q + (size_t)t0 * kDim, kDim, kc * 32);

  v8f o[16];
#pragma unroll
  for (int i = 0; i < 16; ++i) o[i] = {};

  issue(0, 0);
  for (int jb = 0; jb < kSeq / 32; ++jb) {
    int buf = jb & 1;
    if (jb + 1 < kSeq / 32) {
      issue(jb + 1, buf ^ 1);                          // prefetch next tiles
      asm volatile("s_wait_asynccnt 0x10" ::: "memory");  // retire current buffer
    } else {
      asm volatile("s_wait_asynccnt 0x0" ::: "memory");
    }
    __syncthreads();                                   // tiles visible to all waves

    const _Float16* kbuf = kt[buf];
    const _Float16* vbuf = vt[buf];

    v8f s0 = {}, s1 = {};
#pragma unroll
    for (int kc = 0; kc < 8; ++kc) {
      v16h b0 = load_b_p(kbuf, KT_LD, kc * 32);                 // keys 0..15
      v16h b1 = load_b_p(kbuf + 16 * KT_LD, KT_LD, kc * 32);    // keys 16..31
      s0 = wmma_f16(aq[kc], b0, s0);
      s1 = wmma_f16(aq[kc], b1, s1);
    }
    // relu + accumulator layout -> A-operand layout via per-wave LDS tile
#pragma unroll
    for (int g = 0; g < 8; ++g) {
      int m = g + 8 * hi;
      st[wave][m * ST_LD + lx]      = (_Float16)fmaxf(s0[g], 0.0f);
      st[wave][m * ST_LD + 16 + lx] = (_Float16)fmaxf(s1[g], 0.0f);
    }
    v16h sa = load_a_p(st[wave], ST_LD, 0);            // same-wave DS in order
#pragma unroll
    for (int ct = 0; ct < 16; ++ct) {
      v16h bv = load_b_p(vbuf + (ct * 16) * VT_LD, VT_LD, 0);
      o[ct] = wmma_f16(sa, bv, o[ct]);
    }
    __syncthreads();                                   // all waves done reading
  }

  // R = x + O ; store f32 (for final residual/LN) and f16 (FFN GEMM input)
#pragma unroll
  for (int ct = 0; ct < 16; ++ct) {
#pragma unroll
    for (int g = 0; g < 8; ++g) {
      size_t m = (size_t)(t0 + g + 8 * hi);
      int c = ct * 16 + lx;
      float r = x[m * kDim + c] + o[ct][g];
      Rf[m * kDim + c] = r;
      Rh[m * kDim + c] = (_Float16)r;
    }
  }
}

// ------------------------------------------------------------------- launcher
extern "C" void kernel_launch(void* const* d_in, const int* in_sizes, int n_in,
                              void* d_out, int out_size, void* d_ws, size_t ws_size,
                              hipStream_t stream) {
  (void)in_sizes; (void)n_in; (void)out_size; (void)ws_size;
  const float* x       = (const float*)d_in[0];
  const float* norm_g  = (const float*)d_in[1];
  const float* norm_b  = (const float*)d_in[2];
  const float* wq      = (const float*)d_in[3];
  const float* bq      = (const float*)d_in[4];
  const float* wk      = (const float*)d_in[5];
  const float* bk      = (const float*)d_in[6];
  const float* wv      = (const float*)d_in[7];
  const float* bv      = (const float*)d_in[8];
  const float* w1      = (const float*)d_in[9];
  const float* b1      = (const float*)d_in[10];
  const float* w2      = (const float*)d_in[11];
  const float* b2      = (const float*)d_in[12];
  const float* fnorm_g = (const float*)d_in[13];
  const float* fnorm_b = (const float*)d_in[14];

  char* ws = (char*)d_ws;
  const size_t MB = 1024 * 1024;
  _Float16* h    = (_Float16*)(ws + 0);         //  8 MB  [16384,256] f16
  _Float16* Qb   = (_Float16*)(ws + 8  * MB);   //  8 MB  [16384,256] f16 (pre-scaled)
  _Float16* Kb   = (_Float16*)(ws + 16 * MB);   //  8 MB  [16384,256] f16
  _Float16* Vt   = (_Float16*)(ws + 24 * MB);   //  8 MB  [256,16384] f16 (transposed)
  float*    Rf   = (float*)   (ws + 32 * MB);   // 16 MB  [16384,256] f32
  _Float16* Rh   = (_Float16*)(ws + 48 * MB);   //  8 MB  [16384,256] f16
  _Float16* mid  = (_Float16*)(ws + 56 * MB);   // 16 MB  [16384,512] f16
  float*    y    = (float*)   (ws + 72 * MB);   // 16 MB  [16384,256] f32
  _Float16* wqT  = (_Float16*)(ws + 88 * MB);
  _Float16* wkT  = (_Float16*)(ws + 88 * MB + 128 * 1024);
  _Float16* wvT  = (_Float16*)(ws + 88 * MB + 256 * 1024);
  _Float16* w1T  = (_Float16*)(ws + 88 * MB + 384 * 1024);   // [512,256]
  _Float16* w2T  = (_Float16*)(ws + 88 * MB + 640 * 1024);   // [256,512]

  // 1) weights -> transposed f16
  k_transpose_h<<<256, 256, 0, stream>>>(wq, wqT, kDim, kDim);
  k_transpose_h<<<256, 256, 0, stream>>>(wk, wkT, kDim, kDim);
  k_transpose_h<<<256, 256, 0, stream>>>(wv, wvT, kDim, kDim);
  k_transpose_h<<<512, 256, 0, stream>>>(w1, w1T, kDim, kFfn);
  k_transpose_h<<<512, 256, 0, stream>>>(w2, w2T, kFfn, kDim);

  // 2) h = LN(x) -> f16
  k_layernorm<1><<<kTok / 8, 256, 0, stream>>>(x, norm_g, norm_b, h, kTok);

  // 3) Q = (h wq + bq) * DIM^-0.5 ; K = h wk + bk   (f16 out)
  k_gemm_wmma<0><<<dim3(kDim / 16, kTok / 16), 32, 0, stream>>>(
      h, wqT, bq, nullptr, Qb, kDim, kDim, kDim, kDim, 0.0625f);
  k_gemm_wmma<0><<<dim3(kDim / 16, kTok / 16), 32, 0, stream>>>(
      h, wkT, bk, nullptr, Kb, kDim, kDim, kDim, kDim, 1.0f);

  // 4) V^T[d, t] = wv^T h^T + bv  (A = wvT rows over d, B^T = h rows over tokens)
  k_gemm_wmma<2><<<dim3(kTok / 16, kDim / 16), 32, 0, stream>>>(
      wvT, h, bv, nullptr, Vt, kDim, kDim, kDim, kTok, 1.0f);

  // 5) fused attention + residual (async-LDS staged K/V, 4-wave sharing)
  k_attention<<<kTok / (16 * ATTW), 32 * ATTW, 0, stream>>>(Qb, Kb, Vt, x, Rf, Rh);

  // 6) mid = relu(R w1 + b1)  (f16)
  k_gemm_wmma<1><<<dim3(kFfn / 16, kTok / 16), 32, 0, stream>>>(
      Rh, w1T, b1, nullptr, mid, kDim, kDim, kDim, kFfn, 1.0f);

  // 7) y = mid w2 + b2 + R  (f32)
  k_gemm_wmma<3><<<dim3(kDim / 16, kTok / 16), 32, 0, stream>>>(
      mid, w2T, b2, Rf, y, kFfn, kFfn, kFfn, kDim, 1.0f);

  // 8) out = LN(y)
  k_layernorm<0><<<kTok / 8, 256, 0, stream>>>(y, fnorm_g, fnorm_b, d_out, kTok);
}
